// TrafficModelAstgcn_43662637531579
// MI455X (gfx1250) — matrix operand order repaired and checked by reading, not compiled, not run on hardware
//
#include <hip/hip_runtime.h>
#include <hip/hip_bf16.h>
#include <stdint.h>

#define N_NODESC 1000
#define N_EDGESC 16000
#define BATCHC   4
#define T_INC    12
#define F_INC    2
#define K_CHEBC  3
#define C_CHEBC  64
#define C_TIMEC  64
#define N_PREDC  8
#define EPSC     1e-5f
#define E2C      (N_EDGESC + 2 * N_NODESC)   /* 18000 cheb edges incl. self loops */
#define NPADC    1024                        /* padded row count for WMMA operands */
#define KP_ATT   32                          /* K=12 padded to one 16x16x32 step  */
#define KP_SVS   1024                        /* K=1000 padded                     */

typedef __attribute__((ext_vector_type(16))) __bf16  v16bf;
typedef __attribute__((ext_vector_type(8)))  float   v8f;

__device__ __forceinline__ unsigned short f2bf(float f) {
  unsigned int u = __float_as_uint(f);
  unsigned int r = (u + 0x7FFFu + ((u >> 16) & 1u)) >> 16;
  return (unsigned short)r;
}

__device__ __forceinline__ float sigmoidf_(float x) { return 1.0f / (1.0f + __expf(-x)); }

// ---------------------------------------------------------------------------
// bf16 WMMA GEMM, NT form: C[b](MxN) = A[b](MxK) * Bt[b](NxK)^T, f32 accumulate.
// Operands are ZERO-PADDED to NPADC rows x KP cols so the inner loop has no
// guards: per k-step each lane issues 4x global_load_b128 then one
// v_wmma_f32_16x16x32_bf16. 128 threads = 4 waves = 2x2 tiles of 16x16.
// Fragment packing per CDNA5 ISA 7.12.2 (wave32):
//   A 16x32: lane L -> M = L&15; elems 0..7 = K(half*8+0..7), 8..15 = K(16+half*8+0..7)
//   B 32x16: lane L -> N = L&15; elems 0..7 = K(half*8+0..7), 8..15 = K(16+half*8+0..7)
//   C/D f32: elem j -> row = half*8 + j, col = lane&15
// ---------------------------------------------------------------------------
__global__ void k_wmma_gemm_nt(const unsigned short* __restrict__ A,
                               const unsigned short* __restrict__ Bt,
                               float* __restrict__ C,
                               int M, int N, int KP,
                               long long sA, long long sB, long long sC) {
  const int lane = threadIdx.x & 31;
  const int wave = threadIdx.x >> 5;
  const int half = lane >> 4;
  const int r    = lane & 15;
  const int tm   = (blockIdx.y * 2 + (wave >> 1)) * 16;
  const int tn   = (blockIdx.x * 2 + (wave & 1)) * 16;
  const int b    = blockIdx.z;
  const unsigned short* ap = A  + (long long)b * sA + (long long)(tm + r) * KP;
  const unsigned short* bp = Bt + (long long)b * sB + (long long)(tn + r) * KP;
  v8f acc = {};
  for (int k0 = 0; k0 < KP; k0 += 32) {
    union { uint4 q[2]; v16bf h; } fa, fb;
    fa.q[0] = *(const uint4*)(ap + k0 + half * 8);
    fa.q[1] = *(const uint4*)(ap + k0 + 16 + half * 8);
    fb.q[0] = *(const uint4*)(bp + k0 + half * 8);
    fb.q[1] = *(const uint4*)(bp + k0 + 16 + half * 8);
    acc = __builtin_amdgcn_wmma_f32_16x16x32_bf16(false, fa.h, false, fb.h,
                                                  (short)0, acc, false, false);
  }
  float* Cb = C + (long long)b * sC;
  const int bcol = tn + r;
#pragma unroll
  for (int j = 0; j < 8; ++j) {
    int rr = tm + half * 8 + j;
    if (rr < M && bcol < N) Cb[(long long)rr * N + bcol] = acc[j];
  }
}

// pad-convert f32 (R x C) -> bf16 (RP x KP) zero-padded; batch via blockIdx.z
__global__ void k_padcvt_bf16(const float* __restrict__ src, unsigned short* __restrict__ dst,
                              int R, int C, int RP, int KP, long long sSrc, long long sDst) {
  int b = blockIdx.z;
  long long i = (long long)blockIdx.x * blockDim.x + threadIdx.x;
  if (i >= (long long)RP * KP) return;
  int rr = (int)(i / KP), cc = (int)(i % KP);
  unsigned short v = 0;
  if (rr < R && cc < C) v = f2bf(src[(long long)b * sSrc + (long long)rr * C + cc]);
  dst[(long long)b * sDst + i] = v;
}

// transpose-convert f32 (R x C) -> bf16 (NPADC x NPADC) = src^T, zero-padded.
// grid (32,32,B), block (32,8); LDS tile, coalesced both directions.
__global__ void k_transcvt_bf16(const float* __restrict__ src, unsigned short* __restrict__ dst,
                                int R, int C) {
  __shared__ unsigned short tile[32][33];
  int b = blockIdx.z;
  int r0 = blockIdx.y * 32, c0 = blockIdx.x * 32;
  const float* S = src + (long long)b * R * C;
  unsigned short* D = dst + (long long)b * NPADC * NPADC;
  for (int i = (int)threadIdx.y; i < 32; i += 8) {
    int r = r0 + i, c = c0 + (int)threadIdx.x;
    unsigned short v = 0;
    if (r < R && c < C) v = f2bf(S[(long long)r * C + c]);
    tile[i][threadIdx.x] = v;
  }
  __syncthreads();
  for (int i = (int)threadIdx.y; i < 32; i += 8) {
    D[(long long)(c0 + i) * NPADC + r0 + threadIdx.x] = tile[threadIdx.x][i];
  }
}

// ---------------------------------------------------------------------------
// Elementwise / small kernels
// ---------------------------------------------------------------------------
// lhs1[b,t,f] = sum_n X[b,n,f,t] * tU1[n]
__global__ void k_tatt_lhs1(const float* __restrict__ X, const float* __restrict__ tU1,
                            float* __restrict__ lhs1, int F) {
  int i = blockIdx.x * blockDim.x + threadIdx.x;
  if (i >= BATCHC * T_INC * F) return;
  int f = i % F, t = (i / F) % T_INC, b = i / (F * T_INC);
  const float* Xb = X + (long long)b * N_NODESC * F * T_INC;
  float s = 0.f;
  for (int n = 0; n < N_NODESC; ++n) s += Xb[(long long)n * F * T_INC + f * T_INC + t] * tU1[n];
  lhs1[(b * T_INC + t) * F + f] = s;
}

// lhs[b,t,n] = sum_f lhs1[b,t,f] * tU2[f,n]
__global__ void k_tatt_lhs2(const float* __restrict__ lhs1, const float* __restrict__ tU2,
                            float* __restrict__ lhs, int F) {
  int i = blockIdx.x * blockDim.x + threadIdx.x;
  if (i >= BATCHC * T_INC * N_NODESC) return;
  int n = i % N_NODESC, t = (i / N_NODESC) % T_INC, b = i / (N_NODESC * T_INC);
  float s = 0.f;
  for (int f = 0; f < F; ++f) s += lhs1[(b * T_INC + t) * F + f] * tU2[f * N_NODESC + n];
  lhs[(b * T_INC + t) * N_NODESC + n] = s;
}

// rhs[b,n,t] = sum_f tU3[f] * X[b,n,f,t]
__global__ void k_tatt_rhs(const float* __restrict__ X, const float* __restrict__ tU3,
                           float* __restrict__ rhs, int F) {
  int i = blockIdx.x * blockDim.x + threadIdx.x;
  if (i >= BATCHC * N_NODESC * T_INC) return;
  int t = i % T_INC, n = (i / T_INC) % N_NODESC, b = i / (T_INC * N_NODESC);
  const float* Xp = X + ((long long)(b * N_NODESC + n) * F) * T_INC + t;
  float s = 0.f;
  for (int f = 0; f < F; ++f) s += tU3[f] * Xp[(long long)f * T_INC];
  rhs[(b * N_NODESC + n) * T_INC + t] = s;
}

// sig[b,u,s] = sigmoid( sum_n lhs[b,u,n]*rhs[b,n,s] + tbe[u,s] )
__global__ void k_tatt_sig(const float* __restrict__ lhs, const float* __restrict__ rhs,
                           const float* __restrict__ tbe, float* __restrict__ sig) {
  int i = blockIdx.x * blockDim.x + threadIdx.x;
  if (i >= BATCHC * T_INC * T_INC) return;
  int ss = i % T_INC, u = (i / T_INC) % T_INC, b = i / (T_INC * T_INC);
  float s = 0.f;
  for (int n = 0; n < N_NODESC; ++n)
    s += lhs[(b * T_INC + u) * N_NODESC + n] * rhs[(b * N_NODESC + n) * T_INC + ss];
  sig[i] = sigmoidf_(s + tbe[u * T_INC + ss]);
}

// E[b,t,s] = sum_u tVe[t,u] * sig[b,u,s]
__global__ void k_tatt_E(const float* __restrict__ sig, const float* __restrict__ tVe,
                         float* __restrict__ E) {
  int i = blockIdx.x * blockDim.x + threadIdx.x;
  if (i >= BATCHC * T_INC * T_INC) return;
  int ss = i % T_INC, t = (i / T_INC) % T_INC, b = i / (T_INC * T_INC);
  float s = 0.f;
  for (int u = 0; u < T_INC; ++u) s += tVe[t * T_INC + u] * sig[(b * T_INC + u) * T_INC + ss];
  E[i] = s;
}

// softmax over axis 1 (t) of E[b,t,s]
__global__ void k_softmax_axis1_T(const float* __restrict__ E, float* __restrict__ Et) {
  int i = blockIdx.x * blockDim.x + threadIdx.x;
  if (i >= BATCHC * T_INC) return;
  int s = i % T_INC, b = i / T_INC;
  float m = -3.0e38f;
  for (int t = 0; t < T_INC; ++t) m = fmaxf(m, E[(b * T_INC + t) * T_INC + s]);
  float sum = 0.f;
  for (int t = 0; t < T_INC; ++t) sum += __expf(E[(b * T_INC + t) * T_INC + s] - m);
  float inv = 1.f / sum;
  for (int t = 0; t < T_INC; ++t)
    Et[(b * T_INC + t) * T_INC + s] = __expf(E[(b * T_INC + t) * T_INC + s] - m) * inv;
}

// Xt[b,n,f,t] = sum_u X[b,n,f,u] * Et[b,u,t]
__global__ void k_apply_Et(const float* __restrict__ X, const float* __restrict__ Et,
                           float* __restrict__ Xt, int F) {
  int i = blockIdx.x * blockDim.x + threadIdx.x;
  if (i >= BATCHC * N_NODESC * F * T_INC) return;
  int t = i % T_INC, f = (i / T_INC) % F;
  int n = (i / (T_INC * F)) % N_NODESC, b = i / (T_INC * F * N_NODESC);
  const float* Xp = X + (((long long)(b * N_NODESC + n) * F) + f) * T_INC;
  float s = 0.f;
  for (int u = 0; u < T_INC; ++u) s += Xp[u] * Et[(b * T_INC + u) * T_INC + t];
  Xt[i] = s;
}

// l1[b,n,f] = sum_t Xt[b,n,f,t]*sW1[t]
__global__ void k_satt_l1(const float* __restrict__ Xt, const float* __restrict__ sW1,
                          float* __restrict__ l1, int F) {
  int i = blockIdx.x * blockDim.x + threadIdx.x;
  if (i >= BATCHC * N_NODESC * F) return;
  const float* Xp = Xt + (long long)i * T_INC;
  float s = 0.f;
  for (int t = 0; t < T_INC; ++t) s += Xp[t] * sW1[t];
  l1[i] = s;
}

// lhsS[b,n,t] = sum_f l1[b,n,f]*sW2[f,t]
__global__ void k_satt_lhs(const float* __restrict__ l1, const float* __restrict__ sW2,
                           float* __restrict__ lhsS, int F) {
  int i = blockIdx.x * blockDim.x + threadIdx.x;
  if (i >= BATCHC * N_NODESC * T_INC) return;
  int t = i % T_INC; long long bn = i / T_INC;
  float s = 0.f;
  for (int f = 0; f < F; ++f) s += l1[bn * F + f] * sW2[f * T_INC + t];
  lhsS[i] = s;
}

// rhsS[b,n,t] = sum_f sW3[f]*Xt[b,n,f,t]
__global__ void k_satt_rhs(const float* __restrict__ Xt, const float* __restrict__ sW3,
                           float* __restrict__ rhsS, int F) {
  int i = blockIdx.x * blockDim.x + threadIdx.x;
  if (i >= BATCHC * N_NODESC * T_INC) return;
  int t = i % T_INC; long long bn = i / T_INC;
  const float* Xp = Xt + (bn * F) * T_INC + t;
  float s = 0.f;
  for (int f = 0; f < F; ++f) s += sW3[f] * Xp[(long long)f * T_INC];
  rhsS[i] = s;
}

// S0 = sigmoid(prod + sbs) in place
__global__ void k_sigmoid_sbs(float* __restrict__ S0, const float* __restrict__ sbs) {
  long long i = (long long)blockIdx.x * blockDim.x + threadIdx.x;
  if (i >= (long long)BATCHC * N_NODESC * N_NODESC) return;
  long long nm = i % ((long long)N_NODESC * N_NODESC);
  S0[i] = sigmoidf_(S0[i] + sbs[nm]);
}

// column softmax over axis 1 (rows) of S[b,n,k]; in place
__global__ void k_softmax_col(float* __restrict__ S) {
  int k = blockIdx.x, b = blockIdx.y, tid = threadIdx.x;
  float* base = S + (long long)b * N_NODESC * N_NODESC;
  __shared__ float red[256];
  float m = -3.0e38f;
  for (int r = tid; r < N_NODESC; r += 256) m = fmaxf(m, base[(long long)r * N_NODESC + k]);
  red[tid] = m; __syncthreads();
  for (int s = 128; s > 0; s >>= 1) { if (tid < s) red[tid] = fmaxf(red[tid], red[tid + s]); __syncthreads(); }
  m = red[0]; __syncthreads();
  float sum = 0.f;
  for (int r = tid; r < N_NODESC; r += 256) sum += __expf(base[(long long)r * N_NODESC + k] - m);
  red[tid] = sum; __syncthreads();
  for (int s = 128; s > 0; s >>= 1) { if (tid < s) red[tid] += red[tid + s]; __syncthreads(); }
  sum = red[0];
  float inv = 1.f / sum;
  for (int r = tid; r < N_NODESC; r += 256) {
    long long idx = (long long)r * N_NODESC + k;
    base[idx] = __expf(base[idx] - m) * inv;
  }
}

// ----- cheb edge prep -----
__global__ void k_deg_row(const int* __restrict__ row, float* __restrict__ deg) {
  int e = blockIdx.x * blockDim.x + threadIdx.x;
  if (e < N_EDGESC) atomicAdd(&deg[row[e]], 1.0f);
}
__global__ void k_dinv(const float* __restrict__ deg, float* __restrict__ dinv) {
  int n = blockIdx.x * blockDim.x + threadIdx.x;
  if (n < N_NODESC) dinv[n] = deg[n] > 0.f ? rsqrtf(deg[n]) : 0.f;
}
__global__ void k_cheb_build(const int* __restrict__ row, const int* __restrict__ col,
                             const float* __restrict__ dinv,
                             int* __restrict__ rch, int* __restrict__ cch, float* __restrict__ wch) {
  int e = blockIdx.x * blockDim.x + threadIdx.x;
  if (e >= E2C) return;
  if (e < N_EDGESC) { int r = row[e], c = col[e]; rch[e] = r; cch[e] = c; wch[e] = -dinv[r] * dinv[c]; }
  else if (e < N_EDGESC + N_NODESC) { int n = e - N_EDGESC; rch[e] = n; cch[e] = n; wch[e] = 1.f; }
  else { int n = e - N_EDGESC - N_NODESC; rch[e] = n; cch[e] = n; wch[e] = -1.f; }
}

__global__ void k_att_norm(const float* __restrict__ S, const int* __restrict__ rch,
                           const int* __restrict__ cch, const float* __restrict__ wch,
                           float* __restrict__ attn) {
  int i = blockIdx.x * blockDim.x + threadIdx.x;
  if (i >= BATCHC * E2C) return;
  int e = i % E2C, b = i / E2C;
  attn[i] = wch[e] * S[(long long)b * N_NODESC * N_NODESC + (long long)rch[e] * N_NODESC + cch[e]];
}
__global__ void k_diag(const float* __restrict__ S, float* __restrict__ diag) {
  int i = blockIdx.x * blockDim.x + threadIdx.x;
  if (i >= BATCHC * N_NODESC) return;
  int n = i % N_NODESC, b = i / N_NODESC;
  diag[i] = S[(long long)b * N_NODESC * N_NODESC + (long long)n * N_NODESC + n];
}
__global__ void k_scale_diag(const float* __restrict__ X, const float* __restrict__ diag,
                             float* __restrict__ Tx0, int F) {
  int i = blockIdx.x * blockDim.x + threadIdx.x;
  if (i >= BATCHC * N_NODESC * F * T_INC) return;
  int n = (i / (T_INC * F)) % N_NODESC, b = i / (T_INC * F * N_NODESC);
  Tx0[i] = diag[b * N_NODESC + n] * X[i];
}
// out[b,n,c,t] += sum_f Tx[b,n,f,t] * W[f,c]
__global__ void k_addmm(const float* __restrict__ Tx, const float* __restrict__ W,
                        float* __restrict__ out, int F) {
  int i = blockIdx.x * blockDim.x + threadIdx.x;
  if (i >= BATCHC * N_NODESC * C_CHEBC * T_INC) return;
  int t = i % T_INC, c = (i / T_INC) % C_CHEBC;
  long long bn = i / (T_INC * C_CHEBC);
  const float* Txp = Tx + (bn * F) * T_INC + t;
  float s = 0.f;
  for (int f = 0; f < F; ++f) s += Txp[(long long)f * T_INC] * W[f * C_CHEBC + c];
  out[i] += s;
}
// scatter: dst[b, cch[e], f, t] += attn[b,e] * src[b, rch[e], f, t]
__global__ void k_prop(const float* __restrict__ attn, const int* __restrict__ rch,
                       const int* __restrict__ cch, const float* __restrict__ src,
                       float* __restrict__ dst, int F) {
  int i = blockIdx.x * blockDim.x + threadIdx.x;
  if (i >= BATCHC * E2C * T_INC) return;
  int t = i % T_INC, e = (i / T_INC) % E2C, b = i / (T_INC * E2C);
  float a = attn[b * E2C + e];
  const float* sp = src + ((long long)(b * N_NODESC + rch[e]) * F) * T_INC + t;
  float* dp = dst + ((long long)(b * N_NODESC + cch[e]) * F) * T_INC + t;
  for (int f = 0; f < F; ++f) atomicAdd(dp + (long long)f * T_INC, a * sp[(long long)f * T_INC]);
}
__global__ void k_combine_tx2(float* __restrict__ Tx2, const float* __restrict__ Tx0, int n) {
  int i = blockIdx.x * blockDim.x + threadIdx.x;
  if (i < n) Tx2[i] = 2.f * Tx2[i] - Tx0[i];
}
__global__ void k_chb_relu(float* __restrict__ out, const float* __restrict__ chb) {
  int i = blockIdx.x * blockDim.x + threadIdx.x;
  if (i >= BATCHC * N_NODESC * C_CHEBC * T_INC) return;
  int c = (i / T_INC) % C_CHEBC;
  out[i] = fmaxf(out[i] + chb[c], 0.f);
}
// temporal conv: tcO[b,n,co,t] = tcb[co] + sum_ci sum_d Xh[b,n,ci,t+d-1]*tcW[co,ci,d]
__global__ void k_tcconv(const float* __restrict__ Xh, const float* __restrict__ tcW,
                         const float* __restrict__ tcb, float* __restrict__ tcO) {
  int i = blockIdx.x * blockDim.x + threadIdx.x;
  if (i >= BATCHC * N_NODESC * C_TIMEC * T_INC) return;
  int t = i % T_INC, co = (i / T_INC) % C_TIMEC;
  long long bn = i / (T_INC * C_TIMEC);
  const float* Xp = Xh + (bn * C_CHEBC) * T_INC;
  float s = tcb[co];
  for (int ci = 0; ci < C_CHEBC; ++ci) {
#pragma unroll
    for (int d = 0; d < 3; ++d) {
      int ts = t + d - 1;
      if (ts >= 0 && ts < T_INC) s += Xp[(long long)ci * T_INC + ts] * tcW[(co * C_CHEBC + ci) * 3 + d];
    }
  }
  tcO[i] = s;
}
__global__ void k_rcconv(const float* __restrict__ X, const float* __restrict__ rcW,
                         const float* __restrict__ rcb, float* __restrict__ rcO, int F) {
  int i = blockIdx.x * blockDim.x + threadIdx.x;
  if (i >= BATCHC * N_NODESC * C_TIMEC * T_INC) return;
  int t = i % T_INC, co = (i / T_INC) % C_TIMEC;
  long long bn = i / (T_INC * C_TIMEC);
  const float* Xp = X + (bn * F) * T_INC + t;
  float s = rcb[co];
  for (int f = 0; f < F; ++f) s += Xp[(long long)f * T_INC] * rcW[co * F + f];
  rcO[i] = s;
}
// Z = relu(tc+rc); layernorm over channel axis per (b,n,t); out layout (b,n,c,t)
__global__ void k_relu_ln(const float* __restrict__ tcO, const float* __restrict__ rcO,
                          const float* __restrict__ g, const float* __restrict__ be,
                          float* __restrict__ Xout) {
  int i = blockIdx.x * blockDim.x + threadIdx.x;
  if (i >= BATCHC * N_NODESC * T_INC) return;
  long long bn = i / T_INC;
  int t = i % T_INC;
  long long base = (bn * C_TIMEC) * T_INC + t;
  float z[C_TIMEC];
  float m = 0.f;
#pragma unroll
  for (int c = 0; c < C_TIMEC; ++c) {
    float v = fmaxf(tcO[base + (long long)c * T_INC] + rcO[base + (long long)c * T_INC], 0.f);
    z[c] = v; m += v;
  }
  m *= (1.f / C_TIMEC);
  float var = 0.f;
#pragma unroll
  for (int c = 0; c < C_TIMEC; ++c) { float d = z[c] - m; var += d * d; }
  var *= (1.f / C_TIMEC);
  float inv = rsqrtf(var + EPSC);
#pragma unroll
  for (int c = 0; c < C_TIMEC; ++c)
    Xout[base + (long long)c * T_INC] = (z[c] - m) * inv * g[c] + be[c];
}

// ----- final head -----
// Y[b,n,p] = fcb[p] + sum_t sum_c X2[b,n,c,t]*fcW[p,t,0,c]
__global__ void k_fc(const float* __restrict__ X2, const float* __restrict__ fcW,
                     const float* __restrict__ fcb, float* __restrict__ Y) {
  int i = blockIdx.x * blockDim.x + threadIdx.x;
  if (i >= BATCHC * N_NODESC * N_PREDC) return;
  int p = i % N_PREDC;
  long long bn = i / N_PREDC;
  const float* Xp = X2 + (bn * C_TIMEC) * T_INC;
  float s = fcb[p];
  for (int t = 0; t < T_INC; ++t)
    for (int c = 0; c < C_TIMEC; ++c)
      s += Xp[(long long)c * T_INC + t] * fcW[(p * T_INC + t) * C_TIMEC + c];
  Y[i] = s;
}
__global__ void k_degg(const int* __restrict__ col, const float* __restrict__ ew,
                       float* __restrict__ degg) {
  int e = blockIdx.x * blockDim.x + threadIdx.x;
  if (e < N_EDGESC) atomicAdd(&degg[col[e]], ew[e]);
}
__global__ void k_dinvg(const float* __restrict__ degg, float* __restrict__ dinvg) {
  int n = blockIdx.x * blockDim.x + threadIdx.x;
  if (n < N_NODESC) dinvg[n] = rsqrtf(degg[n] + 1.0f);  // +1 for self loop weight
}
__global__ void k_mm_small(const float* __restrict__ Xin, const float* __restrict__ W,
                           float* __restrict__ out, int FI, int FO) {
  int i = blockIdx.x * blockDim.x + threadIdx.x;
  if (i >= BATCHC * N_NODESC * FO) return;
  int fo = i % FO;
  long long bn = i / FO;
  float s = 0.f;
  for (int fi = 0; fi < FI; ++fi) s += Xin[bn * FI + fi] * W[fi * FO + fo];
  out[i] = s;
}
__global__ void k_gcn_scatter(const int* __restrict__ row, const int* __restrict__ col,
                              const float* __restrict__ ew, const float* __restrict__ dinvg,
                              const float* __restrict__ h0, float* __restrict__ out, int FO) {
  int i = blockIdx.x * blockDim.x + threadIdx.x;
  if (i >= BATCHC * N_EDGESC) return;
  int e = i % N_EDGESC, b = i / N_EDGESC;
  int r = row[e], c = col[e];
  float nrm = dinvg[r] * ew[e] * dinvg[c];
  const float* sp = h0 + (long long)(b * N_NODESC + r) * FO;
  float* dp = out + (long long)(b * N_NODESC + c) * FO;
  for (int f = 0; f < FO; ++f) atomicAdd(&dp[f], nrm * sp[f]);
}
__global__ void k_gcn_self_bias(const float* __restrict__ dinvg, const float* __restrict__ h0,
                                const float* __restrict__ bias, float* __restrict__ out, int FO) {
  int i = blockIdx.x * blockDim.x + threadIdx.x;
  if (i >= BATCHC * N_NODESC * FO) return;
  int f = i % FO, n = (i / FO) % N_NODESC;
  out[i] += dinvg[n] * dinvg[n] * h0[i] + bias[f];
}
__global__ void k_ln2_stats(const float* __restrict__ h, float* __restrict__ stats) {
  __shared__ float s1[256]; __shared__ float s2[256];
  int b = blockIdx.x, tid = threadIdx.x;
  float a = 0.f, q = 0.f;
  for (int i = tid; i < N_NODESC * 16; i += 256) {
    float v = h[(long long)b * N_NODESC * 16 + i];
    a += v; q += v * v;
  }
  s1[tid] = a; s2[tid] = q; __syncthreads();
  for (int s = 128; s > 0; s >>= 1) {
    if (tid < s) { s1[tid] += s1[tid + s]; s2[tid] += s2[tid + s]; }
    __syncthreads();
  }
  if (tid == 0) { stats[b * 2] = s1[0]; stats[b * 2 + 1] = s2[0]; }
}
__global__ void k_ln2_apply(const float* __restrict__ h1, const float* __restrict__ stats,
                            const float* __restrict__ g, const float* __restrict__ be,
                            float* __restrict__ out) {
  int i = blockIdx.x * blockDim.x + threadIdx.x;
  if (i >= BATCHC * N_NODESC * 16) return;
  int b = i / (N_NODESC * 16), nf = i % (N_NODESC * 16);
  const float cnt = (float)(N_NODESC * 16);
  float m = stats[b * 2] / cnt;
  float v = stats[b * 2 + 1] / cnt - m * m;
  out[i] = (h1[i] - m) * rsqrtf(v + EPSC) * g[nf] + be[nf];
}
__global__ void k_final(const float* __restrict__ h2, const float* __restrict__ fW,
                        const float* __restrict__ fb, float* __restrict__ out) {
  int i = blockIdx.x * blockDim.x + threadIdx.x;
  if (i >= BATCHC * N_NODESC) return;
  float s = fb[0];
  for (int k = 0; k < 8; ++k) s += h2[(long long)i * 8 + k] * fW[k];
  out[i] = s;
}

// ---------------------------------------------------------------------------
// Host orchestration
// ---------------------------------------------------------------------------
struct BlkP {
  const float *chW, *chb, *lnb, *lng, *rcW, *rcb, *sVs, *sW1, *sW2, *sW3, *sbs,
              *tU1, *tU2, *tU3, *tVe, *tbe, *tcW, *tcb;
};

struct WsB {
  float *lhs1, *lhsT, *rhsT, *sig, *Ebuf, *Et, *Xt, *l1, *lhsS, *rhsS, *S0, *S1;
  float *attn, *diag, *Tx0, *Tx1, *Tx2, *outC, *tcO, *rcO;
  unsigned short *sVsbf, *S0bfT, *lhsSbf, *rhsSbf;
  int *rch, *cch;
  float *wch, *deg, *dinv, *degg, *dinvg;
  float *Xb1, *Xb2, *Y, *h0, *h1, *hn, *h2a, *h2, *stats;
};

#define L(kern, n, ...) kern<<<dim3((unsigned)(((n) + 255) / 256)), dim3(256), 0, stream>>>(__VA_ARGS__)

static void run_block(hipStream_t stream, WsB& W, const float* Xin, int F,
                      const BlkP& p, float* Xout) {
  const long long NN2 = (long long)N_NODESC * N_NODESC;
  // --- temporal attention ---
  L(k_tatt_lhs1, BATCHC * T_INC * F, Xin, p.tU1, W.lhs1, F);
  L(k_tatt_lhs2, BATCHC * T_INC * N_NODESC, W.lhs1, p.tU2, W.lhsT, F);
  L(k_tatt_rhs, BATCHC * N_NODESC * T_INC, Xin, p.tU3, W.rhsT, F);
  L(k_tatt_sig, BATCHC * T_INC * T_INC, W.lhsT, W.rhsT, p.tbe, W.sig);
  L(k_tatt_E, BATCHC * T_INC * T_INC, W.sig, p.tVe, W.Ebuf);
  L(k_softmax_axis1_T, BATCHC * T_INC, W.Ebuf, W.Et);
  L(k_apply_Et, BATCHC * N_NODESC * F * T_INC, Xin, W.Et, W.Xt, F);
  // --- spatial attention ---
  L(k_satt_l1, BATCHC * N_NODESC * F, W.Xt, p.sW1, W.l1, F);
  L(k_satt_lhs, BATCHC * N_NODESC * T_INC, W.l1, p.sW2, W.lhsS, F);
  L(k_satt_rhs, BATCHC * N_NODESC * T_INC, W.Xt, p.sW3, W.rhsS, F);
  // pad-convert both (N x 12) operands into (1024 x 32) zero-padded bf16
  {
    dim3 g((unsigned)((NPADC * KP_ATT + 255) / 256), 1, BATCHC);
    k_padcvt_bf16<<<g, dim3(256), 0, stream>>>(W.lhsS, W.lhsSbf, N_NODESC, T_INC, NPADC, KP_ATT,
                                               (long long)N_NODESC * T_INC, (long long)NPADC * KP_ATT);
    k_padcvt_bf16<<<g, dim3(256), 0, stream>>>(W.rhsS, W.rhsSbf, N_NODESC, T_INC, NPADC, KP_ATT,
                                               (long long)N_NODESC * T_INC, (long long)NPADC * KP_ATT);
  }
  dim3 gg(NPADC / 32, NPADC / 32, BATCHC);   // 2x2 16x16 tiles per 128-thread block
  // prod[b] = lhsS (N x 12) * rhsS^T   [WMMA bf16, unguarded b128 fragment loads]
  k_wmma_gemm_nt<<<gg, dim3(128), 0, stream>>>(
      W.lhsSbf, W.rhsSbf, W.S0, N_NODESC, N_NODESC, KP_ATT,
      (long long)NPADC * KP_ATT, (long long)NPADC * KP_ATT, NN2);
  L(k_sigmoid_sbs, (int)(BATCHC * NN2), W.S0, p.sbs);
  // A = sVs pad-converted (1024 x 1024); B^T = S0 transposed via LDS tiles
  {
    dim3 g((unsigned)(((long long)NPADC * KP_SVS + 255) / 256), 1, 1);
    k_padcvt_bf16<<<g, dim3(256), 0, stream>>>(p.sVs, W.sVsbf, N_NODESC, N_NODESC, NPADC, KP_SVS,
                                               0LL, 0LL);
  }
  k_transcvt_bf16<<<dim3(NPADC / 32, NPADC / 32, BATCHC), dim3(32, 8), 0, stream>>>(
      W.S0, W.S0bfT, N_NODESC, N_NODESC);
  // S1[b] = sVs (N x N) * S0[b]   [WMMA bf16, K=1000 padded to 1024]
  k_wmma_gemm_nt<<<gg, dim3(128), 0, stream>>>(
      W.sVsbf, W.S0bfT, W.S1, N_NODESC, N_NODESC, KP_SVS,
      0LL, (long long)NPADC * KP_SVS, NN2);
  k_softmax_col<<<dim3(N_NODESC, BATCHC), dim3(256), 0, stream>>>(W.S1);
  // --- chebyshev graph conv (vmapped over t via index math) ---
  L(k_att_norm, BATCHC * E2C, W.S1, W.rch, W.cch, W.wch, W.attn);
  L(k_diag, BATCHC * N_NODESC, W.S1, W.diag);
  size_t txBytes = (size_t)BATCHC * N_NODESC * F * T_INC * sizeof(float);
  size_t outBytes = (size_t)BATCHC * N_NODESC * C_CHEBC * T_INC * sizeof(float);
  L(k_scale_diag, BATCHC * N_NODESC * F * T_INC, Xin, W.diag, W.Tx0, F);
  hipMemsetAsync(W.outC, 0, outBytes, stream);
  L(k_addmm, BATCHC * N_NODESC * C_CHEBC * T_INC, W.Tx0, p.chW, W.outC, F);
  hipMemsetAsync(W.Tx1, 0, txBytes, stream);
  L(k_prop, BATCHC * E2C * T_INC, W.attn, W.rch, W.cch, W.Tx0, W.Tx1, F);
  L(k_addmm, BATCHC * N_NODESC * C_CHEBC * T_INC, W.Tx1, p.chW + 1 * F * C_CHEBC, W.outC, F);
  hipMemsetAsync(W.Tx2, 0, txBytes, stream);
  L(k_prop, BATCHC * E2C * T_INC, W.attn, W.rch, W.cch, W.Tx1, W.Tx2, F);
  L(k_combine_tx2, BATCHC * N_NODESC * F * T_INC, W.Tx2, W.Tx0, BATCHC * N_NODESC * F * T_INC);
  L(k_addmm, BATCHC * N_NODESC * C_CHEBC * T_INC, W.Tx2, p.chW + 2 * F * C_CHEBC, W.outC, F);
  L(k_chb_relu, BATCHC * N_NODESC * C_CHEBC * T_INC, W.outC, p.chb);
  // --- temporal conv + residual + layernorm ---
  L(k_tcconv, BATCHC * N_NODESC * C_TIMEC * T_INC, W.outC, p.tcW, p.tcb, W.tcO);
  L(k_rcconv, BATCHC * N_NODESC * C_TIMEC * T_INC, Xin, p.rcW, p.rcb, W.rcO, F);
  L(k_relu_ln, BATCHC * N_NODESC * T_INC, W.tcO, W.rcO, p.lng, p.lnb, Xout);
}

extern "C" void kernel_launch(void* const* d_in, const int* in_sizes, int n_in,
                              void* d_out, int out_size, void* d_ws, size_t ws_size,
                              hipStream_t stream) {
  (void)in_sizes; (void)n_in; (void)out_size; (void)ws_size;

  // Inputs (JAX pytree order: dict keys sorted, ASCII / uppercase-first).
  const float* x   = (const float*)d_in[0];
  const int* eidx  = (const int*)d_in[1];
  const float* ewt = (const float*)d_in[2];
  const int* row = eidx;
  const int* col = eidx + N_EDGESC;

  int idx = 3;
  BlkP blk[2];
  for (int bI = 0; bI < 2; ++bI) {
    BlkP& p = blk[bI];
    p.chW = (const float*)d_in[idx++]; p.chb = (const float*)d_in[idx++];
    p.lnb = (const float*)d_in[idx++]; p.lng = (const float*)d_in[idx++];
    p.rcW = (const float*)d_in[idx++]; p.rcb = (const float*)d_in[idx++];
    p.sVs = (const float*)d_in[idx++]; p.sW1 = (const float*)d_in[idx++];
    p.sW2 = (const float*)d_in[idx++]; p.sW3 = (const float*)d_in[idx++];
    p.sbs = (const float*)d_in[idx++]; p.tU1 = (const float*)d_in[idx++];
    p.tU2 = (const float*)d_in[idx++]; p.tU3 = (const float*)d_in[idx++];
    p.tVe = (const float*)d_in[idx++]; p.tbe = (const float*)d_in[idx++];
    p.tcW = (const float*)d_in[idx++]; p.tcb = (const float*)d_in[idx++];
  }
  const float* fW   = (const float*)d_in[idx++];
  const float* fb   = (const float*)d_in[idx++];
  const float* fcW  = (const float*)d_in[idx++];
  const float* fcb  = (const float*)d_in[idx++];
  const float* g1W  = (const float*)d_in[idx++];
  const float* g1b  = (const float*)d_in[idx++];
  const float* g2W  = (const float*)d_in[idx++];
  const float* g2b  = (const float*)d_in[idx++];
  const float* olnb = (const float*)d_in[idx++];
  const float* olng = (const float*)d_in[idx++];

  // Workspace bump allocation
  char* wp = (char*)d_ws;
  auto alloc = [&](size_t bytes) -> void* {
    void* r = (void*)wp;
    wp += (bytes + 255) & ~(size_t)255;
    return r;
  };
  const size_t fBNCT = (size_t)BATCHC * N_NODESC * C_TIMEC * T_INC;
  const size_t fBNN  = (size_t)BATCHC * N_NODESC * N_NODESC;

  WsB W;
  W.Xb1  = (float*)alloc(fBNCT * 4);
  W.Xb2  = (float*)alloc(fBNCT * 4);
  W.Xt   = (float*)alloc(fBNCT * 4);
  W.Tx0  = (float*)alloc(fBNCT * 4);
  W.Tx1  = (float*)alloc(fBNCT * 4);
  W.Tx2  = (float*)alloc(fBNCT * 4);
  W.outC = (float*)alloc(fBNCT * 4);
  W.tcO  = (float*)alloc(fBNCT * 4);
  W.rcO  = (float*)alloc(fBNCT * 4);
  W.S0   = (float*)alloc(fBNN * 4);
  W.S1   = (float*)alloc(fBNN * 4);
  W.S0bfT  = (unsigned short*)alloc((size_t)BATCHC * NPADC * KP_SVS * 2);
  W.sVsbf  = (unsigned short*)alloc((size_t)NPADC * KP_SVS * 2);
  W.lhsSbf = (unsigned short*)alloc((size_t)BATCHC * NPADC * KP_ATT * 2);
  W.rhsSbf = (unsigned short*)alloc((size_t)BATCHC * NPADC * KP_ATT * 2);
  W.lhs1 = (float*)alloc((size_t)BATCHC * T_INC * C_TIMEC * 4);
  W.lhsT = (float*)alloc((size_t)BATCHC * T_INC * N_NODESC * 4);
  W.rhsT = (float*)alloc((size_t)BATCHC * N_NODESC * T_INC * 4);
  W.lhsS = (float*)alloc((size_t)BATCHC * N_NODESC * T_INC * 4);
  W.rhsS = (float*)alloc((size_t)BATCHC * N_NODESC * T_INC * 4);
  W.sig  = (float*)alloc((size_t)BATCHC * T_INC * T_INC * 4);
  W.Ebuf = (float*)alloc((size_t)BATCHC * T_INC * T_INC * 4);
  W.Et   = (float*)alloc((size_t)BATCHC * T_INC * T_INC * 4);
  W.l1   = (float*)alloc((size_t)BATCHC * N_NODESC * C_TIMEC * 4);
  W.attn = (float*)alloc((size_t)BATCHC * E2C * 4);
  W.diag = (float*)alloc((size_t)BATCHC * N_NODESC * 4);
  W.rch  = (int*)alloc((size_t)E2C * 4);
  W.cch  = (int*)alloc((size_t)E2C * 4);
  W.wch  = (float*)alloc((size_t)E2C * 4);
  W.deg  = (float*)alloc((size_t)N_NODESC * 4);
  W.dinv = (float*)alloc((size_t)N_NODESC * 4);
  W.degg = (float*)alloc((size_t)N_NODESC * 4);
  W.dinvg = (float*)alloc((size_t)N_NODESC * 4);
  W.Y    = (float*)alloc((size_t)BATCHC * N_NODESC * N_PREDC * 4);
  W.h0   = (float*)alloc((size_t)BATCHC * N_NODESC * 16 * 4);
  W.h1   = (float*)alloc((size_t)BATCHC * N_NODESC * 16 * 4);
  W.hn   = (float*)alloc((size_t)BATCHC * N_NODESC * 16 * 4);
  W.h2a  = (float*)alloc((size_t)BATCHC * N_NODESC * 8 * 4);
  W.h2   = (float*)alloc((size_t)BATCHC * N_NODESC * 8 * 4);
  W.stats = (float*)alloc(8 * 4);

  // --- Chebyshev edge prep (shared by both blocks) ---
  hipMemsetAsync(W.deg, 0, N_NODESC * 4, stream);
  L(k_deg_row, N_EDGESC, row, W.deg);
  L(k_dinv, N_NODESC, W.deg, W.dinv);
  L(k_cheb_build, E2C, row, col, W.dinv, W.rch, W.cch, W.wch);

  // --- two ASTGCN blocks ---
  run_block(stream, W, x, F_INC, blk[0], W.Xb1);
  run_block(stream, W, W.Xb1, C_TIMEC, blk[1], W.Xb2);

  // --- final FC over (t, c) ---
  L(k_fc, BATCHC * N_NODESC * N_PREDC, W.Xb2, fcW, fcb, W.Y);

  // --- GCN norms (edge weights + self loops) ---
  hipMemsetAsync(W.degg, 0, N_NODESC * 4, stream);
  L(k_degg, N_EDGESC, col, ewt, W.degg);
  L(k_dinvg, N_NODESC, W.degg, W.dinvg);

  // --- GCN 1: 8 -> 16, then LN over (N,16) per batch ---
  L(k_mm_small, BATCHC * N_NODESC * 16, W.Y, g1W, W.h0, N_PREDC, 16);
  hipMemsetAsync(W.h1, 0, (size_t)BATCHC * N_NODESC * 16 * 4, stream);
  L(k_gcn_scatter, BATCHC * N_EDGESC, row, col, ewt, W.dinvg, W.h0, W.h1, 16);
  L(k_gcn_self_bias, BATCHC * N_NODESC * 16, W.dinvg, W.h0, g1b, W.h1, 16);
  k_ln2_stats<<<dim3(BATCHC), dim3(256), 0, stream>>>(W.h1, W.stats);
  L(k_ln2_apply, BATCHC * N_NODESC * 16, W.h1, W.stats, olng, olnb, W.hn);

  // --- GCN 2: 16 -> 8 ---
  L(k_mm_small, BATCHC * N_NODESC * 8, W.hn, g2W, W.h2a, 16, 8);
  hipMemsetAsync(W.h2, 0, (size_t)BATCHC * N_NODESC * 8 * 4, stream);
  L(k_gcn_scatter, BATCHC * N_EDGESC, row, col, ewt, W.dinvg, W.h2a, W.h2, 8);
  L(k_gcn_self_bias, BATCHC * N_NODESC * 8, W.dinvg, W.h2a, g2b, W.h2, 8);

  // --- final linear: (B,N,8) @ (8,1) + fb ---
  L(k_final, BATCHC * N_NODESC, W.h2, fW, fb, (float*)d_out);
}
